// BERT_27444841022096
// MI455X (gfx1250) — matrix operand order repaired and checked by reading, not compiled, not run on hardware
//
#include <hip/hip_runtime.h>
#include <hip/hip_bf16.h>

typedef __attribute__((ext_vector_type(16))) _Float16 v16h;
typedef __attribute__((ext_vector_type(8)))  _Float16 v8h;
typedef __attribute__((ext_vector_type(8)))  float    v8f;

#define TOK    4096
#define DMODEL 768
#define NHEAD  12
#define DHEAD  64
#define NLAYER 12
#define DFF    3072
#define NVOCAB 32000
#define NB     8
#define NSEQ   512
#define LN10K  9.210340371976184f

__device__ __forceinline__ float gelu_exact(float x) {
  return 0.5f * x * (1.0f + erff(x * 0.70710678118654752f));
}

__device__ __forceinline__ float block_reduce_sum(float v, float* sh) {
  int t = threadIdx.x;
  sh[t] = v; __syncthreads();
  for (int s = 128; s > 0; s >>= 1) {
    if (t < s) sh[t] += sh[t + s];
    __syncthreads();
  }
  float r = sh[0]; __syncthreads();
  return r;
}

__device__ __forceinline__ float block_reduce_max(float v, float* sh) {
  int t = threadIdx.x;
  sh[t] = v; __syncthreads();
  for (int s = 128; s > 0; s >>= 1) {
    if (t < s) sh[t] = fmaxf(sh[t], sh[t + s]);
    __syncthreads();
  }
  float r = sh[0]; __syncthreads();
  return r;
}

__device__ __forceinline__ v16h cat16(v8h a, v8h b) {
  return __builtin_shufflevector(a, b, 0,1,2,3,4,5,6,7,8,9,10,11,12,13,14,15);
}

// ---------------------------------------------------------------------------
// Generic batched WMMA GEMM: C[z] = A[z] (MxK f16) * B[z] (KxN f16) [+bias]
// Block: 256 threads (8 waves). Block tile 128x64, wave tile 32x32 (4 WMMA).
// B tile double-buffered in LDS (transposed [n][k]) -> 1 barrier per k-step.
// C offset for batch z: (z/zdivC)*sC1 + (z%zdivC)*sC2  (head-merge trick).
// ---------------------------------------------------------------------------
template<bool GELU>
__global__ __launch_bounds__(256)
void wmma_gemm(const _Float16* __restrict__ A, long sA, int lda,
               const _Float16* __restrict__ Bw, long sB, int ldb,
               const float* __restrict__ bias,
               float* __restrict__ C32, _Float16* __restrict__ C16,
               int zdivC, long sC1, long sC2, int ldc,
               int K)
{
  __shared__ __align__(16) _Float16 Bs[2][64][40];  // [buf][n][k], pad 8 halves

  const int z = blockIdx.z;
  const _Float16* Ab = A  + (long)z * sA;
  const _Float16* Bb = Bw + (long)z * sB;
  const long cOff = (long)(z / zdivC) * sC1 + (long)(z % zdivC) * sC2;

  const int mbase = blockIdx.y * 128;
  const int nbase = blockIdx.x * 64;
  const int tid  = threadIdx.x;
  const int wave = tid >> 5;
  const int lane = tid & 31;
  const int wm = (wave & 3) * 32;          // wave row offset in block tile
  const int wn = (wave >> 2) * 32;         // wave col offset in block tile
  const int lm = lane & 15;
  const int hi = (lane & 16) ? 1 : 0;      // half-wave select
  const int koff = hi ? 8 : 0;             // A fragment K offset per ISA layout
  const int bk   = hi ? 8 : 0;             // B fragment K offset per ISA layout

  // cooperative B stage indices: coalesced 16B global loads
  const int kk  = tid >> 3;                // 0..31
  const int nn0 = (tid & 7) << 3;          // 0..56

  v8f c00 = {}, c01 = {}, c10 = {}, c11 = {};
  const _Float16* Arow0 = Ab + (long)(mbase + wm + lm) * lda;
  const _Float16* Arow1 = Arow0 + 16l * lda;

  // ---- stage B tile [32k x 64n] for k0 into LDS buffer p, transposed ----
  auto stage = [&](int k0, int p) {
    const _Float16* gp = Bb + (long)(k0 + kk) * ldb + nbase + nn0;
    v8h bv = *(const v8h*)gp;
    if (k0 + 32 < K)
      __builtin_prefetch(gp + 32l * ldb, 0, 1);   // global_prefetch_b8
    #pragma unroll
    for (int i = 0; i < 8; ++i) Bs[p][nn0 + i][kk] = bv[i];
  };

  stage(0, 0);
  __syncthreads();

  int p = 0;
  for (int k0 = 0; k0 < K; k0 += 32) {
    if (k0 + 32 < K) stage(k0 + 32, p ^ 1);   // fill other buffer

    // ---- A fragments straight from global (16x32 f16 layout) ----
    const _Float16* ap0 = Arow0 + k0 + koff;
    const _Float16* ap1 = Arow1 + k0 + koff;
    v16h a0 = cat16(*(const v8h*)(ap0), *(const v8h*)(ap0 + 16));
    v16h a1 = cat16(*(const v8h*)(ap1), *(const v8h*)(ap1 + 16));

    // ---- B fragments from LDS (column = lane, K = bk..bk+7 and bk+16..bk+23) ----
    const v8h* bp0 = (const v8h*)&Bs[p][wn + lm][bk];
    const v8h* bp1 = (const v8h*)&Bs[p][wn + 16 + lm][bk];
    v16h b0 = cat16(bp0[0], bp0[2]);
    v16h b1 = cat16(bp1[0], bp1[2]);

    c00 = __builtin_amdgcn_wmma_f32_16x16x32_f16(false, a0, false, b0,
                                                 (short)0, c00, false, false);
    c01 = __builtin_amdgcn_wmma_f32_16x16x32_f16(false, a0, false, b1,
                                                 (short)0, c01, false, false);
    c10 = __builtin_amdgcn_wmma_f32_16x16x32_f16(false, a1, false, b0,
                                                 (short)0, c10, false, false);
    c11 = __builtin_amdgcn_wmma_f32_16x16x32_f16(false, a1, false, b1,
                                                 (short)0, c11, false, false);

    __syncthreads();   // staging of p^1 done; compute on p done before reuse
    p ^= 1;
  }

  // ---- epilogue: C/D layout VGPR r -> row r (+8 for upper half-wave) ----
  const int row0 = mbase + wm + hi * 8;
  const int col0 = nbase + wn + lm;
  float bia0 = bias ? bias[col0]      : 0.0f;
  float bia1 = bias ? bias[col0 + 16] : 0.0f;
  #pragma unroll
  for (int r = 0; r < 8; ++r) {
    float v00 = c00[r] + bia0, v01 = c01[r] + bia1;
    float v10 = c10[r] + bia0, v11 = c11[r] + bia1;
    if (GELU) {
      v00 = gelu_exact(v00); v01 = gelu_exact(v01);
      v10 = gelu_exact(v10); v11 = gelu_exact(v11);
    }
    long base0 = cOff + (long)(row0 + r) * ldc;
    long base1 = cOff + (long)(row0 + 16 + r) * ldc;
    if (C32) {
      C32[base0 + col0] = v00; C32[base0 + col0 + 16] = v01;
      C32[base1 + col0] = v10; C32[base1 + col0 + 16] = v11;
    }
    if (C16) {
      C16[base0 + col0] = (_Float16)v00; C16[base0 + col0 + 16] = (_Float16)v01;
      C16[base1 + col0] = (_Float16)v10; C16[base1 + col0 + 16] = (_Float16)v11;
    }
  }
}

// ---------------------------------------------------------------------------
// Embedding gather + sinusoidal PE + LayerNorm -> h32, h16
// ---------------------------------------------------------------------------
__global__ __launch_bounds__(256)
void embed_ln(const float* __restrict__ tok_emb, const int* __restrict__ x,
              const float* __restrict__ g, const float* __restrict__ bb,
              float* __restrict__ h32, _Float16* __restrict__ h16)
{
  __shared__ float sh[256];
  const long t = blockIdx.x;            // token
  const int n = (int)(t & (NSEQ - 1));  // position in sequence
  const int id = x[t];
  const float* e = tok_emb + (long)id * DMODEL;

  float xv[3], s = 0.f, q = 0.f;
  #pragma unroll
  for (int i = 0; i < 3; ++i) {
    int d = threadIdx.x + 256 * i;
    int de = d & ~1;
    float ang = (float)n * __expf(-(float)de * (LN10K / (float)DMODEL));
    float pe = (d & 1) ? __cosf(ang) : __sinf(ang);
    float v = e[d] + pe;
    xv[i] = v; s += v; q += v * v;
  }
  s = block_reduce_sum(s, sh);
  q = block_reduce_sum(q, sh);
  float mu = s * (1.f / DMODEL);
  float rstd = rsqrtf(q * (1.f / DMODEL) - mu * mu + 1e-5f);
  #pragma unroll
  for (int i = 0; i < 3; ++i) {
    int d = threadIdx.x + 256 * i;
    float y = (xv[i] - mu) * rstd * g[d] + bb[d];
    h32[t * DMODEL + d] = y;
    h16[t * DMODEL + d] = (_Float16)y;
  }
}

// ---------------------------------------------------------------------------
// Residual add + LN (g1,b1) -> out32; optional second LN (g2,b2) -> out16
// ---------------------------------------------------------------------------
__global__ __launch_bounds__(256)
void residual_ln(const float* __restrict__ a, const float* __restrict__ res,
                 const float* __restrict__ g1, const float* __restrict__ b1,
                 const float* __restrict__ g2, const float* __restrict__ b2,
                 float* __restrict__ out32, _Float16* __restrict__ out16)
{
  __shared__ float sh[256];
  const long t = blockIdx.x;
  const float* ap = a + t * DMODEL;
  const float* rp = res ? res + t * DMODEL : nullptr;

  float xv[3], s = 0.f, q = 0.f;
  #pragma unroll
  for (int i = 0; i < 3; ++i) {
    int d = threadIdx.x + 256 * i;
    float v = ap[d] + (rp ? rp[d] : 0.f);
    xv[i] = v; s += v; q += v * v;
  }
  s = block_reduce_sum(s, sh);
  q = block_reduce_sum(q, sh);
  float mu = s * (1.f / DMODEL);
  float rstd = rsqrtf(q * (1.f / DMODEL) - mu * mu + 1e-5f);

  float yv[3];
  #pragma unroll
  for (int i = 0; i < 3; ++i) {
    int d = threadIdx.x + 256 * i;
    yv[i] = (xv[i] - mu) * rstd * g1[d] + b1[d];
    if (out32) out32[t * DMODEL + d] = yv[i];
  }
  if (g2) {
    float s2 = 0.f, q2 = 0.f;
    #pragma unroll
    for (int i = 0; i < 3; ++i) { s2 += yv[i]; q2 += yv[i] * yv[i]; }
    s2 = block_reduce_sum(s2, sh);
    q2 = block_reduce_sum(q2, sh);
    float mu2 = s2 * (1.f / DMODEL);
    float r2 = rsqrtf(q2 * (1.f / DMODEL) - mu2 * mu2 + 1e-5f);
    #pragma unroll
    for (int i = 0; i < 3; ++i) {
      int d = threadIdx.x + 256 * i;
      out16[t * DMODEL + d] = (_Float16)((yv[i] - mu2) * r2 * g2[d] + b2[d]);
    }
  } else if (out16) {
    #pragma unroll
    for (int i = 0; i < 3; ++i) {
      int d = threadIdx.x + 256 * i;
      out16[t * DMODEL + d] = (_Float16)yv[i];
    }
  }
}

// ---------------------------------------------------------------------------
// RoPE (first 32 dims/head) + split QKV into head-major f16:
//   q16[b,h,n,d], kT16[b,h,d,n] (pre-transposed for Q*K^T), v16[b,h,n,d]
// ---------------------------------------------------------------------------
__global__ __launch_bounds__(256)
void rope_split(const float* __restrict__ qkv,
                _Float16* __restrict__ q16, _Float16* __restrict__ kT16,
                _Float16* __restrict__ v16)
{
  const long t = blockIdx.x;
  const int b = (int)(t >> 9);
  const int n = (int)(t & (NSEQ - 1));
  const float* base = qkv + t * (3 * DMODEL);
  for (int e = threadIdx.x; e < DMODEL; e += 256) {
    int hh = e >> 6, d = e & 63;
    float qv = base[hh * DHEAD + d];
    float kv = base[DMODEL + hh * DHEAD + d];
    float vv = base[2 * DMODEL + hh * DHEAD + d];
    if (d < 32) {
      int de = d & ~1;
      float inv = __expf(-(float)de * (LN10K / 32.0f));
      float ang = (float)n * inv;
      float c = __cosf(ang), s = __sinf(ang);
      int dp = d ^ 1;
      float qp = base[hh * DHEAD + dp];
      float kp = base[DMODEL + hh * DHEAD + dp];
      float qr = (d & 1) ? qp : -qp;
      float kr = (d & 1) ? kp : -kp;
      qv = qv * c + qr * s;
      kv = kv * c + kr * s;
    }
    long hb = (long)b * NHEAD + hh;
    q16[(hb * NSEQ + n) * DHEAD + d] = (_Float16)qv;
    kT16[(hb * DHEAD + d) * NSEQ + n] = (_Float16)kv;
    v16[(hb * NSEQ + n) * DHEAD + d] = (_Float16)vv;
  }
}

// ---------------------------------------------------------------------------
// scale + pad-mask + softmax over key dim (512), write f16 probabilities
// ---------------------------------------------------------------------------
__global__ __launch_bounds__(256)
void softmax_mask(const float* __restrict__ scores, const int* __restrict__ x,
                  _Float16* __restrict__ attn)
{
  __shared__ float sh[256];
  const long row = blockIdx.x;                    // over B*H*N
  const int b = (int)(row / (NHEAD * NSEQ));
  const float* s = scores + row * NSEQ;
  _Float16* o = attn + row * NSEQ;
  const int* xb = x + (long)b * NSEQ;
  const int t = threadIdx.x;

  float v[2], mx = -1e30f;
  #pragma unroll
  for (int i = 0; i < 2; ++i) {
    int j = t + i * 256;
    float val = (xb[j] == 0) ? 1e-9f : s[j] * 0.125f;   // SCALE = 64^-0.5
    v[i] = val; mx = fmaxf(mx, val);
  }
  mx = block_reduce_max(mx, sh);
  float e0 = __expf(v[0] - mx), e1 = __expf(v[1] - mx);
  float sum = block_reduce_sum(e0 + e1, sh);
  float inv = 1.0f / sum;
  o[t] = (_Float16)(e0 * inv);
  o[t + 256] = (_Float16)(e1 * inv);
}

__global__ void cvt16(const float* __restrict__ in, _Float16* __restrict__ out, long n)
{
  long i = (long)blockIdx.x * 256 + threadIdx.x;
  long stride = (long)gridDim.x * 256;
  for (; i < n; i += stride) out[i] = (_Float16)in[i];
}

__global__ void copy_h0(const float* __restrict__ h32, float* __restrict__ out)
{
  int b = blockIdx.x;
  for (int d = threadIdx.x; d < DMODEL; d += 256)
    out[(long)b * DMODEL + d] = h32[((long)b * NSEQ) * (long)DMODEL + d];
}

// ---------------------------------------------------------------------------
extern "C" void kernel_launch(void* const* d_in, const int* in_sizes, int n_in,
                              void* d_out, int out_size, void* d_ws, size_t ws_size,
                              hipStream_t stream)
{
  const float* tok_emb = (const float*)d_in[0];
  const float* emb_g = (const float*)d_in[1];
  const float* emb_b = (const float*)d_in[2];
  const float* n1_g = (const float*)d_in[3];
  const float* n1_b = (const float*)d_in[4];
  const float* n2_g = (const float*)d_in[5];
  const float* n2_b = (const float*)d_in[6];
  const float* qkv_w = (const float*)d_in[7];
  const float* out_w = (const float*)d_in[8];
  const float* ff_g = (const float*)d_in[9];
  const float* ff_b = (const float*)d_in[10];
  const float* ff_w1 = (const float*)d_in[11];
  const float* ff_b1 = (const float*)d_in[12];
  const float* ff_w2 = (const float*)d_in[13];
  const float* ff_b2 = (const float*)d_in[14];
  const float* h_w1 = (const float*)d_in[15];
  const float* h_b1 = (const float*)d_in[16];
  const float* h_g = (const float*)d_in[17];
  const float* h_b = (const float*)d_in[18];
  const float* h_w2 = (const float*)d_in[19];
  const float* h_b2 = (const float*)d_in[20];
  const int*   x = (const int*)d_in[21];

  // ---- workspace layout ----
  char* cur = (char*)d_ws;
  auto alloc = [&](long elems, int esz) -> char* {
    char* p = cur;
    cur += (((long)elems * esz) + 255) & ~255l;
    return p;
  };
  _Float16* qkv16  = (_Float16*)alloc((long)NLAYER * DMODEL * 3 * DMODEL, 2);
  _Float16* out16w = (_Float16*)alloc((long)NLAYER * DMODEL * DMODEL, 2);
  _Float16* ffw116 = (_Float16*)alloc((long)NLAYER * DMODEL * DFF, 2);
  _Float16* ffw216 = (_Float16*)alloc((long)NLAYER * DFF * DMODEL, 2);
  _Float16* hw116  = (_Float16*)alloc((long)DMODEL * DMODEL, 2);
  _Float16* hw216  = (_Float16*)alloc((long)DMODEL * NVOCAB, 2);
  float*    h32    = (float*)   alloc((long)TOK * DMODEL, 4);
  _Float16* h16    = (_Float16*)alloc((long)TOK * DMODEL, 2);
  float*    qkv32  = (float*)   alloc((long)TOK * 3 * DMODEL, 4);
  _Float16* q16    = (_Float16*)alloc((long)NB * NHEAD * NSEQ * DHEAD, 2);
  _Float16* kT16   = (_Float16*)alloc((long)NB * NHEAD * NSEQ * DHEAD, 2);
  _Float16* v16p   = (_Float16*)alloc((long)NB * NHEAD * NSEQ * DHEAD, 2);
  float*    sc32   = (float*)   alloc((long)NB * NHEAD * NSEQ * NSEQ, 4);
  _Float16* attn16 = (_Float16*)alloc((long)NB * NHEAD * NSEQ * NSEQ, 2);
  _Float16* o16    = (_Float16*)alloc((long)TOK * DMODEL, 2);
  _Float16* t116   = (_Float16*)alloc((long)TOK * DFF, 2);
  float*    f32buf = (float*)   alloc((long)TOK * DMODEL, 4);
  _Float16* head16 = (_Float16*)alloc((long)TOK * DMODEL, 2);

  auto gemm = [&](const _Float16* A, long sA, int lda,
                  const _Float16* Bm, long sB, int ldb,
                  const float* bias, float* C32, _Float16* C16,
                  int zdiv, long sC1, long sC2, int ldc,
                  int Mg, int Ng, int Kg, int batch, bool gelu) {
    dim3 grid(Ng / 64, Mg / 128, batch);
    if (gelu)
      wmma_gemm<true><<<grid, 256, 0, stream>>>(A, sA, lda, Bm, sB, ldb, bias,
                                                C32, C16, zdiv, sC1, sC2, ldc, Kg);
    else
      wmma_gemm<false><<<grid, 256, 0, stream>>>(A, sA, lda, Bm, sB, ldb, bias,
                                                 C32, C16, zdiv, sC1, sC2, ldc, Kg);
  };

  // ---- weights f32 -> f16 (every call; deterministic) ----
  cvt16<<<4096, 256, 0, stream>>>(qkv_w, qkv16, (long)NLAYER * DMODEL * 3 * DMODEL);
  cvt16<<<4096, 256, 0, stream>>>(out_w, out16w, (long)NLAYER * DMODEL * DMODEL);
  cvt16<<<4096, 256, 0, stream>>>(ff_w1, ffw116, (long)NLAYER * DMODEL * DFF);
  cvt16<<<4096, 256, 0, stream>>>(ff_w2, ffw216, (long)NLAYER * DFF * DMODEL);
  cvt16<<<1024, 256, 0, stream>>>(h_w1, hw116, (long)DMODEL * DMODEL);
  cvt16<<<4096, 256, 0, stream>>>(h_w2, hw216, (long)DMODEL * NVOCAB);

  // ---- embedding + PE + LN ----
  embed_ln<<<TOK, 256, 0, stream>>>(tok_emb, x, emb_g, emb_b, h32, h16);

  // ---- transformer layers ----
  for (int l = 0; l < NLAYER; ++l) {
    const _Float16* wq = qkv16 + (long)l * DMODEL * 3 * DMODEL;
    const _Float16* wo = out16w + (long)l * DMODEL * DMODEL;
    const _Float16* w1 = ffw116 + (long)l * DMODEL * DFF;
    const _Float16* w2 = ffw216 + (long)l * DFF * DMODEL;

    // qkv = h @ w_qkv
    gemm(h16, 0, DMODEL, wq, 0, 3 * DMODEL, nullptr, qkv32, nullptr,
         1, (long)TOK * 3 * DMODEL, 0, 3 * DMODEL, TOK, 3 * DMODEL, DMODEL, 1, false);
    // rope + head split
    rope_split<<<TOK, 256, 0, stream>>>(qkv32, q16, kT16, v16p);
    // scores = q @ k^T  (96 batches of 512x512x64)
    gemm(q16, (long)NSEQ * DHEAD, DHEAD, kT16, (long)DHEAD * NSEQ, NSEQ, nullptr,
         sc32, nullptr, 1, (long)NSEQ * NSEQ, 0, NSEQ,
         NSEQ, NSEQ, DHEAD, NB * NHEAD, false);
    // masked softmax -> f16
    softmax_mask<<<NB * NHEAD * NSEQ, 256, 0, stream>>>(sc32, x, attn16);
    // o = attn @ v, heads merged into [B,N,H*DH] via zdiv offset trick
    gemm(attn16, (long)NSEQ * NSEQ, NSEQ, v16p, (long)NSEQ * DHEAD, DHEAD, nullptr,
         nullptr, o16, NHEAD, (long)NSEQ * DMODEL, DHEAD, DMODEL,
         NSEQ, DHEAD, NSEQ, NB * NHEAD, false);
    // out projection
    gemm(o16, 0, DMODEL, wo, 0, DMODEL, nullptr, f32buf, nullptr,
         1, (long)TOK * DMODEL, 0, DMODEL, TOK, DMODEL, DMODEL, 1, false);
    // h = LN(o + h); h16 = LN(h, ff_g, ff_b) for FF input
    residual_ln<<<TOK, 256, 0, stream>>>(f32buf, h32, n1_g, n1_b,
                                         ff_g + (long)l * DMODEL, ff_b + (long)l * DMODEL,
                                         h32, h16);
    // f1 = gelu(h16 @ w1 + b1) -> f16
    gemm(h16, 0, DMODEL, w1, 0, DFF, ff_b1 + (long)l * DFF, nullptr, t116,
         1, (long)TOK * DFF, 0, DFF, TOK, DFF, DMODEL, 1, true);
    // f = f1 @ w2 + b2
    gemm(t116, 0, DFF, w2, 0, DMODEL, ff_b2 + (long)l * DMODEL, f32buf, nullptr,
         1, (long)TOK * DMODEL, 0, DMODEL, TOK, DMODEL, DFF, 1, false);
    // h = LN(f + h); h16 = f16(h) for next layer
    residual_ln<<<TOK, 256, 0, stream>>>(f32buf, h32, n2_g, n2_b,
                                         nullptr, nullptr, h32, h16);
  }

  // ---- head ----
  // gelu(h @ h_w1 + h_b1)
  gemm(h16, 0, DMODEL, hw116, 0, DMODEL, h_b1, f32buf, nullptr,
       1, (long)TOK * DMODEL, 0, DMODEL, TOK, DMODEL, DMODEL, 1, true);
  // head16 = LN(., h_g, h_b)
  residual_ln<<<TOK, 256, 0, stream>>>(f32buf, nullptr, h_g, h_b,
                                       nullptr, nullptr, nullptr, head16);
  // logits = head @ h_w2 + h_b2  -> d_out (f32)
  gemm(head16, 0, DMODEL, hw216, 0, NVOCAB, h_b2, (float*)d_out, nullptr,
       1, (long)TOK * NVOCAB, 0, NVOCAB, TOK, NVOCAB, DMODEL, 1, false);
  // h[:,0] -> d_out tail
  copy_h0<<<NB, 256, 0, stream>>>(h32, (float*)d_out + (long)NB * NSEQ * NVOCAB);
}
